// MinimalAttnHead_55980603736552
// MI455X (gfx1250) — matrix operand order repaired and checked by reading, not compiled, not run on hardware
//
#include <hip/hip_runtime.h>

// ---------------------------------------------------------------------------
// MinimalAttnHead on MI455X (gfx1250, wave32)
//   K  = E @ W^T                      (WMMA f32_16x16x32_f16 GEMM)
//   Et = E^T (fp16, per batch)        (LDS tile transpose)
//   out = softmax(causal(E K^T /16)) @ E
//       flash attention: TDM double-buffered K/V staging + WMMA + online softmax
// Workspace: Kf16 [B*S*D] + Etf16 [B*D*S] + Wf16 [D*D]  (~16.9 MB)
// ---------------------------------------------------------------------------

typedef __attribute__((ext_vector_type(16))) _Float16 v16h;
typedef __attribute__((ext_vector_type(8)))  _Float16 v8h;
typedef __attribute__((ext_vector_type(8)))  float    v8f;
typedef __attribute__((ext_vector_type(4)))  unsigned int u32x4;
typedef __attribute__((ext_vector_type(8)))  int      i32x8;
typedef __attribute__((ext_vector_type(4)))  int      i32x4;

#define B_N 4
#define S_N 4096
#define D_N 256
#define SCALE_F 0.0625f      // 1/sqrt(256)
#define NEG_F  (-1.0e9f)

__device__ __forceinline__ v8f wmma_f16(v16h a, v16h b, v8f c) {
  return __builtin_amdgcn_wmma_f32_16x16x32_f16(false, a, false, b, (short)0, c,
                                                false, false);
}

// --- Tensor Data Mover: 2D fp16 tile (tile_w elems/row, tile_h rows) --------
// D# packing per CDNA5 ISA (groups 0/1; 2D -> groups 2,3 = 0)
// 6-arg builtin form (clang-23 / therock-10.0 headers).
__device__ __forceinline__ void tdm_load_2d_f16(unsigned lds_off, const void* gptr,
                                                unsigned tile_w, unsigned tile_h,
                                                unsigned stride_elems) {
  const unsigned long long ga = (unsigned long long)gptr;
  u32x4 g0;
  g0[0] = 1u;                                            // count=1 (user descriptor)
  g0[1] = lds_off;                                       // lds_addr (bytes)
  g0[2] = (unsigned)ga;                                  // global_addr[31:0]
  g0[3] = (unsigned)((ga >> 32) & 0x01FFFFFFu) | 0x80000000u; // addr[56:32] | type=2
  i32x8 g1;
  g1[0] = 1 << 16;                                       // wg_mask=0, data_size=1 (2B)
  g1[1] = (int)((tile_w & 0xFFFFu) << 16);               // tensor_dim0[15:0]
  g1[2] = (int)((tile_w >> 16) | ((tile_h & 0xFFFFu) << 16)); // dim0 hi | dim1 lo
  g1[3] = (int)((tile_h >> 16) | ((tile_w & 0xFFFFu) << 16)); // dim1 hi | tile_dim0
  g1[4] = (int)(tile_h & 0xFFFFu);                       // tile_dim1 | tile_dim2=0
  g1[5] = (int)stride_elems;                             // tensor_dim0_stride[31:0]
  g1[6] = 0;                                             // stride0 hi | stride1 lo
  g1[7] = 0;                                             // stride1 hi
  const i32x4 z4 = {0, 0, 0, 0};
  const i32x8 z8 = {0, 0, 0, 0, 0, 0, 0, 0};
  __builtin_amdgcn_tensor_load_to_lds(g0, g1, z4, z4, z8, 0);
}

// ------------------------- kernel 1: W fp32 -> fp16 -------------------------
__global__ void cvt_w_f16(const float* __restrict__ W, _Float16* __restrict__ Wh) {
  int i = blockIdx.x * blockDim.x + threadIdx.x;   // exactly D*D threads
  Wh[i] = (_Float16)W[i];
}

// ------------- kernel 2: Et[b][d][s] = (fp16) E[b][s][d] --------------------
// 64x64 tiles through LDS; grid = B * (S/64) * (D/64) = 1024 blocks x 256 thr
__global__ __launch_bounds__(256, 1)
void transpose_e(const float* __restrict__ E, _Float16* __restrict__ Et) {
  __shared__ _Float16 T[64][72];                   // padded vs bank conflicts
  const int tid = threadIdx.x;
  const int b   = blockIdx.x >> 8;
  const int rem = blockIdx.x & 255;
  const int s0  = (rem >> 2) * 64;
  const int d0  = (rem & 3) * 64;
  const float*  Eb  = E  + (size_t)b * S_N * D_N;
  _Float16*     Etb = Et + (size_t)b * D_N * S_N;
#pragma unroll
  for (int j = 0; j < 16; ++j) {
    const int idx = j * 256 + tid;
    const int si = idx >> 6, di = idx & 63;
    T[di][si] = (_Float16)Eb[(size_t)(s0 + si) * D_N + d0 + di];
  }
  __syncthreads();
#pragma unroll
  for (int j = 0; j < 16; ++j) {
    const int idx = j * 256 + tid;
    const int di = idx >> 6, si = idx & 63;
    Etb[(size_t)(d0 + di) * S_N + s0 + si] = T[di][si];
  }
}

// -------------------- kernel 3: K = E @ W^T  (fp16 out) ---------------------
__global__ __launch_bounds__(128, 1)
void proj_k(const float* __restrict__ E, const _Float16* __restrict__ Wh,
            _Float16* __restrict__ Kh) {
  const int lane = threadIdx.x & 31;
  const int w    = threadIdx.x >> 5;
  const int n16  = lane & 15;
  const int hi   = lane >> 4;
  const int row0 = (blockIdx.x * 4 + w) * 16;

  const float* erow = E + (size_t)(row0 + n16) * D_N;
  v16h a[8];
#pragma unroll
  for (int c = 0; c < 8; ++c) {
    const float4 f0 = *(const float4*)(erow + 32 * c + 16 * hi);
    const float4 f1 = *(const float4*)(erow + 32 * c + 16 * hi + 4);
    const float4 f2 = *(const float4*)(erow + 32 * c + 16 * hi + 8);
    const float4 f3 = *(const float4*)(erow + 32 * c + 16 * hi + 12);
    a[c][0]=(_Float16)f0.x; a[c][1]=(_Float16)f0.y; a[c][2]=(_Float16)f0.z; a[c][3]=(_Float16)f0.w;
    a[c][4]=(_Float16)f1.x; a[c][5]=(_Float16)f1.y; a[c][6]=(_Float16)f1.z; a[c][7]=(_Float16)f1.w;
    a[c][8]=(_Float16)f2.x; a[c][9]=(_Float16)f2.y; a[c][10]=(_Float16)f2.z; a[c][11]=(_Float16)f2.w;
    a[c][12]=(_Float16)f3.x; a[c][13]=(_Float16)f3.y; a[c][14]=(_Float16)f3.z; a[c][15]=(_Float16)f3.w;
  }

#pragma unroll
  for (int t = 0; t < 16; ++t) {
    v8f acc = {};
#pragma unroll
    for (int c = 0; c < 8; ++c) {
      v16h bw = *(const v16h*)(Wh + (size_t)(16 * t + n16) * D_N + 32 * c + 16 * hi);
      acc = wmma_f16(a[c], bw, acc);
    }
#pragma unroll
    for (int r = 0; r < 8; ++r)
      Kh[(size_t)(row0 + r + 8 * hi) * D_N + 16 * t + n16] = (_Float16)acc[r];
  }
}

// ----------------- kernel 4: causal flash attention -------------------------
// grid = B * (S/64), 128 threads (4 waves). Wave w owns query rows
// [q0wg + 16w, +16). 32-key blocks staged by TDM, double-buffered.
__global__ __launch_bounds__(128, 1)
void flash_attn(const float* __restrict__ E, const _Float16* __restrict__ Kh,
                const _Float16* __restrict__ Et, float* __restrict__ Out) {
  __shared__ __align__(16) _Float16 Kl[2][32][256];   // K block [key][d]
  __shared__ __align__(16) _Float16 Vl[2][256][32];   // V block transposed [d][key]
  __shared__ __align__(16) _Float16 Pl[4][16][32];    // per-wave P scratch

  const int tid  = threadIdx.x;
  const int lane = tid & 31;
  const int w    = tid >> 5;
  const int n16  = lane & 15;
  const int hi   = lane >> 4;

  const int b    = blockIdx.x >> 6;
  const int qb   = blockIdx.x & 63;
  const int q0wg = qb * 64;
  const int q0w  = q0wg + w * 16;

  const float*    Eb  = E  + (size_t)b * S_N * D_N;
  const _Float16* Kb  = Kh + (size_t)b * S_N * D_N;
  const _Float16* Etb = Et + (size_t)b * D_N * S_N;

  // ---- Q fragment (A layout), fp32 -> fp16, registers ----------------------
  v16h qa[8];
  {
    const float* qrow = Eb + (size_t)(q0w + n16) * D_N;
#pragma unroll
    for (int c = 0; c < 8; ++c) {
      const float4 f0 = *(const float4*)(qrow + 32 * c + 16 * hi);
      const float4 f1 = *(const float4*)(qrow + 32 * c + 16 * hi + 4);
      const float4 f2 = *(const float4*)(qrow + 32 * c + 16 * hi + 8);
      const float4 f3 = *(const float4*)(qrow + 32 * c + 16 * hi + 12);
      qa[c][0]=(_Float16)f0.x; qa[c][1]=(_Float16)f0.y; qa[c][2]=(_Float16)f0.z; qa[c][3]=(_Float16)f0.w;
      qa[c][4]=(_Float16)f1.x; qa[c][5]=(_Float16)f1.y; qa[c][6]=(_Float16)f1.z; qa[c][7]=(_Float16)f1.w;
      qa[c][8]=(_Float16)f2.x; qa[c][9]=(_Float16)f2.y; qa[c][10]=(_Float16)f2.z; qa[c][11]=(_Float16)f2.w;
      qa[c][12]=(_Float16)f3.x; qa[c][13]=(_Float16)f3.y; qa[c][14]=(_Float16)f3.z; qa[c][15]=(_Float16)f3.w;
    }
  }

  v8f o[16];
  {
    v8f z = {};
#pragma unroll
    for (int t = 0; t < 16; ++t) o[t] = z;
  }
  float m[8], l[8];
#pragma unroll
  for (int r = 0; r < 8; ++r) { m[r] = -3.0e38f; l[r] = 0.0f; }

  const int nkb = qb * 2 + 2;            // keys [0, q0wg+64)

  // ---- prologue: TDM-stage block 0 into buffer 0 ---------------------------
  if (w == 0) {
    tdm_load_2d_f16((unsigned)(size_t)&Kl[0][0][0], Kb,  256, 32, 256);
    tdm_load_2d_f16((unsigned)(size_t)&Vl[0][0][0], Etb, 32, 256, S_N);
  }

  for (int kbi = 0; kbi < nkb; ++kbi) {
    const int kb  = kbi * 32;
    const int buf = kbi & 1;
    if (w == 0) __builtin_amdgcn_s_wait_tensorcnt(0);  // buffer `buf` landed
    __syncthreads();                                   // visible to all waves

    // issue TDM for next block into the other buffer (overlaps compute)
    if (w == 0 && (kbi + 1) < nkb) {
      const int kn = (kbi + 1) * 32;
      tdm_load_2d_f16((unsigned)(size_t)&Kl[buf ^ 1][0][0],
                      Kb + (size_t)kn * D_N, 256, 32, 256);
      tdm_load_2d_f16((unsigned)(size_t)&Vl[buf ^ 1][0][0],
                      Etb + kn, 32, 256, S_N);
    }

    // ---- S = Q * K^T (two 16x16 tiles over 32 keys) ------------------------
    v8f s0 = {}, s1 = {};
#pragma unroll
    for (int c = 0; c < 8; ++c) {
      v16h b0 = *(const v16h*)(&Kl[buf][n16][32 * c + 16 * hi]);
      s0 = wmma_f16(qa[c], b0, s0);
      v16h b1 = *(const v16h*)(&Kl[buf][16 + n16][32 * c + 16 * hi]);
      s1 = wmma_f16(qa[c], b1, s1);
    }

    // ---- scale, causal mask, online softmax --------------------------------
    float al[8];
#pragma unroll
    for (int r = 0; r < 8; ++r) {
      const int q = q0w + r + 8 * hi;
      float x0 = s0[r] * SCALE_F; if (kb + n16 > q)      x0 = NEG_F;
      float x1 = s1[r] * SCALE_F; if (kb + 16 + n16 > q) x1 = NEG_F;
      float mb = fmaxf(x0, x1);
      mb = fmaxf(mb, __shfl_xor(mb, 1, 32));
      mb = fmaxf(mb, __shfl_xor(mb, 2, 32));
      mb = fmaxf(mb, __shfl_xor(mb, 4, 32));
      mb = fmaxf(mb, __shfl_xor(mb, 8, 32));
      const float mn = fmaxf(m[r], mb);
      const float a  = __expf(m[r] - mn);
      m[r] = mn;
      const float p0 = __expf(x0 - mn);
      const float p1 = __expf(x1 - mn);
      float rs = p0 + p1;
      rs += __shfl_xor(rs, 1, 32);
      rs += __shfl_xor(rs, 2, 32);
      rs += __shfl_xor(rs, 4, 32);
      rs += __shfl_xor(rs, 8, 32);
      l[r]  = l[r] * a + rs;
      al[r] = a;
      Pl[w][r + 8 * hi][n16]      = (_Float16)p0;   // C layout -> row-major
      Pl[w][r + 8 * hi][16 + n16] = (_Float16)p1;
    }
#pragma unroll
    for (int t = 0; t < 16; ++t) {
#pragma unroll
      for (int r = 0; r < 8; ++r) o[t][r] *= al[r];
    }

    __syncthreads();                     // P visible across the wave's lanes

    // ---- O += P * V --------------------------------------------------------
    v8h plo = *(const v8h*)(&Pl[w][n16][16 * hi]);
    v8h phi = *(const v8h*)(&Pl[w][n16][16 * hi + 8]);
    v16h pa = __builtin_shufflevector(plo, phi, 0, 1, 2, 3, 4, 5, 6, 7,
                                      8, 9, 10, 11, 12, 13, 14, 15);
#pragma unroll
    for (int t = 0; t < 16; ++t) {
      v16h vb = *(const v16h*)(&Vl[buf][16 * t + n16][16 * hi]);
      o[t] = wmma_f16(pa, vb, o[t]);
    }
  }

  // ---- normalize and write out ---------------------------------------------
  float* ob = Out + (size_t)b * S_N * D_N;
#pragma unroll
  for (int r = 0; r < 8; ++r) {
    const float inv = 1.0f / l[r];
    float* orow = ob + (size_t)(q0w + r + 8 * hi) * D_N;
#pragma unroll
    for (int t = 0; t < 16; ++t) orow[16 * t + n16] = o[t][r] * inv;
  }
}

// ---------------------------------------------------------------------------
extern "C" void kernel_launch(void* const* d_in, const int* in_sizes, int n_in,
                              void* d_out, int out_size, void* d_ws, size_t ws_size,
                              hipStream_t stream) {
  (void)in_sizes; (void)n_in; (void)out_size; (void)ws_size;
  const float* E = (const float*)d_in[0];   // [B,S,D] fp32
  const float* W = (const float*)d_in[1];   // [D,D]  fp32
  float* out = (float*)d_out;               // [B,S,D] fp32

  char* ws = (char*)d_ws;
  _Float16* Kh = (_Float16*)ws;                                       // B*S*D
  _Float16* Et = (_Float16*)(ws + (size_t)B_N * S_N * D_N * 2);       // B*D*S
  _Float16* Wh = (_Float16*)(ws + (size_t)2 * B_N * S_N * D_N * 2);   // D*D

  cvt_w_f16<<<(D_N * D_N) / 256, 256, 0, stream>>>(W, Wh);
  transpose_e<<<B_N * (S_N / 64) * (D_N / 64), 256, 0, stream>>>(E, Et);
  proj_k<<<(B_N * S_N) / 64, 128, 0, stream>>>(E, Wh, Kh);
  flash_attn<<<B_N * (S_N / 64), 128, 0, stream>>>(E, Kh, Et, out);
}
